// InbetweenerTM_89103391523642
// MI455X (gfx1250) — compile-verified, matrix operand order checked by reading, not run on hardware
//
#include <hip/hip_runtime.h>
#include <math.h>

typedef __attribute__((ext_vector_type(16))) _Float16 v16h;
typedef __attribute__((ext_vector_type(8)))  _Float16 h8;
typedef __attribute__((ext_vector_type(8)))  float    v8f;
typedef __attribute__((ext_vector_type(4)))  float    f4;

#define NB    2
#define NPTS  1536
#define DIMC  128
#define NHEAD 4
#define DHEAD 32
#define M1    1537

__device__ __forceinline__ v8f wmma_f16(v16h a, v16h b, v8f c) {
  // D(16x16 f32) = A(16x32 f16) * B(32x16 f16) + C
  return __builtin_amdgcn_wmma_f32_16x16x32_f16(false, a, false, b, (short)0, c, false, false);
}

// Fragment from two contiguous 8-half runs (lo: K=g*8+e, hi: K=16+g*8+e)
__device__ __forceinline__ v16h frag_from(const _Float16* p) {
  h8 lo = *(const h8*)(p);
  h8 hi = *(const h8*)(p + 16);
  v16h f;
#pragma unroll
  for (int i = 0; i < 8; ++i) { f[i] = lo[i]; f[8 + i] = hi[i]; }
  return f;
}

// A fragment from an f32 weight row (two contiguous float4 pairs -> f16)
__device__ __forceinline__ v16h frag_from_f32(const float* p) {
  f4 w0 = *(const f4*)(p);
  f4 w1 = *(const f4*)(p + 4);
  f4 w2 = *(const f4*)(p + 16);
  f4 w3 = *(const f4*)(p + 20);
  v16h a;
#pragma unroll
  for (int i = 0; i < 4; ++i) {
    a[i] = (_Float16)w0[i]; a[4 + i] = (_Float16)w1[i];
    a[8 + i] = (_Float16)w2[i]; a[12 + i] = (_Float16)w3[i];
  }
  return a;
}

// ---------------------------------------------------------------------------
// 1x1 Conv GEMM, activations [B,N,C]: Y[b,n,co] = sum_ci X[b,n,ci]*W[co,ci]+b
// Cin % 32 == 0. One wave -> 16(co) x 64(n) tile, software-pipelined K loop.
// ---------------------------------------------------------------------------
__global__ void k_conv(const _Float16* __restrict__ X, const float* __restrict__ W,
                       const float* __restrict__ bias, float* __restrict__ Y32,
                       _Float16* __restrict__ Y16, int Cin, int Cout) {
  const int lane = threadIdx.x;
  const int g = lane >> 4, r = lane & 15, c = lane & 15;
  const int n0 = blockIdx.x * 64, co0 = blockIdx.y * 16, b = blockIdx.z;
  v8f acc0 = {}, acc1 = {}, acc2 = {}, acc3 = {};
  const float* wrow = W + (size_t)(co0 + r) * Cin + g * 8;
  const _Float16* x0p = X + (size_t)(b * NPTS + n0 + c) * Cin + g * 8;
  const _Float16* x1p = x0p + (size_t)16 * Cin;
  const _Float16* x2p = x0p + (size_t)32 * Cin;
  const _Float16* x3p = x0p + (size_t)48 * Cin;

  v16h a  = frag_from_f32(wrow);
  v16h b0 = frag_from(x0p);
  v16h b1 = frag_from(x1p);
  v16h b2 = frag_from(x2p);
  v16h b3 = frag_from(x3p);
  int k0 = 0;
  for (;;) {
    int kn = k0 + 32;
    if (kn >= Cin) {
      acc0 = wmma_f16(a, b0, acc0);
      acc1 = wmma_f16(a, b1, acc1);
      acc2 = wmma_f16(a, b2, acc2);
      acc3 = wmma_f16(a, b3, acc3);
      break;
    }
    // prefetch next chunk's fragments before issuing current WMMAs
    v16h an  = frag_from_f32(wrow + kn);
    v16h nb0 = frag_from(x0p + kn);
    v16h nb1 = frag_from(x1p + kn);
    v16h nb2 = frag_from(x2p + kn);
    v16h nb3 = frag_from(x3p + kn);
    acc0 = wmma_f16(a, b0, acc0);
    acc1 = wmma_f16(a, b1, acc1);
    acc2 = wmma_f16(a, b2, acc2);
    acc3 = wmma_f16(a, b3, acc3);
    a = an; b0 = nb0; b1 = nb1; b2 = nb2; b3 = nb3; k0 = kn;
  }
  f4 bs0 = *(const f4*)(bias + co0 + g * 8);
  f4 bs1 = *(const f4*)(bias + co0 + g * 8 + 4);
  v8f accs[4] = {acc0, acc1, acc2, acc3};
#pragma unroll
  for (int j = 0; j < 4; ++j) {
    size_t row = (size_t)(b * NPTS + n0 + j * 16 + c) * Cout + co0 + g * 8;
    f4 o0, o1; h8 o16;
#pragma unroll
    for (int v = 0; v < 4; ++v) {
      o0[v] = accs[j][v] + bs0[v];
      o1[v] = accs[j][4 + v] + bs1[v];
      o16[v] = (_Float16)o0[v];
      o16[4 + v] = (_Float16)o1[v];
    }
    *(f4*)(Y32 + row) = o0;
    *(f4*)(Y32 + row + 4) = o1;
    *(h8*)(Y16 + row) = o16;
  }
}

// ---------------------------------------------------------------------------
// Special conv for Cin=2 (kpts encoder head) reading raw f32 [B,N,2]
// ---------------------------------------------------------------------------
__global__ void k_conv2(const float* __restrict__ X, const float* __restrict__ W,
                        const float* __restrict__ bias, float* __restrict__ Y32,
                        _Float16* __restrict__ Y16) {
  int idx = blockIdx.x * blockDim.x + threadIdx.x;
  int total = NB * NPTS * 32;
  if (idx >= total) return;
  int co = idx & 31;
  int bn = idx >> 5;
  float x0 = X[bn * 2], x1 = X[bn * 2 + 1];
  float y = fmaf(W[co * 2], x0, fmaf(W[co * 2 + 1], x1, bias[co]));
  Y32[idx] = y;
  Y16[idx] = (_Float16)y;
}

// ---------------------------------------------------------------------------
// InstanceNorm1d(affine=False)+ReLU over N for each (b,c); [B,N,C] layout
// ---------------------------------------------------------------------------
__global__ void k_inorm_relu(const float* __restrict__ X, _Float16* __restrict__ Y, int C) {
  const int cch = blockIdx.x % C, b = blockIdx.x / C;
  const float* x = X + (size_t)b * NPTS * C + cch;
  _Float16* y = Y + (size_t)b * NPTS * C + cch;
  __shared__ float s1[256], s2[256];
  float sum = 0.f, sq = 0.f;
  for (int i = threadIdx.x; i < NPTS; i += 256) { float t = x[(size_t)i * C]; sum += t; sq += t * t; }
  s1[threadIdx.x] = sum; s2[threadIdx.x] = sq; __syncthreads();
  for (int off = 128; off > 0; off >>= 1) {
    if (threadIdx.x < off) { s1[threadIdx.x] += s1[threadIdx.x + off]; s2[threadIdx.x] += s2[threadIdx.x + off]; }
    __syncthreads();
  }
  float mean = s1[0] * (1.0f / NPTS);
  float var  = s2[0] * (1.0f / NPTS) - mean * mean;
  float rstd = rsqrtf(var + 1e-5f);
  for (int i = threadIdx.x; i < NPTS; i += 256) {
    float t = (x[(size_t)i * C] - mean) * rstd;
    y[(size_t)i * C] = (_Float16)(t > 0.f ? t : 0.f);
  }
}

__global__ void k_cvt(const float* __restrict__ X, _Float16* __restrict__ Y, int count) {
  int idx = blockIdx.x * blockDim.x + threadIdx.x;
  if (idx < count) Y[idx] = (_Float16)X[idx];
}

// y32 += x32 ; y16 = (f16)y32
__global__ void k_addpack(float* __restrict__ y32, const float* __restrict__ x32,
                          _Float16* __restrict__ y16, int count) {
  int idx = blockIdx.x * blockDim.x + threadIdx.x;
  if (idx >= count) return;
  float t = y32[idx] + x32[idx];
  y32[idx] = t;
  y16[idx] = (_Float16)t;
}

// [B,N,128] f32 (ch = d*4+h) -> [B,H,N,32] f16
__global__ void k_packqk(const float* __restrict__ X, _Float16* __restrict__ Y) {
  int idx = blockIdx.x * blockDim.x + threadIdx.x;
  int total = NB * NHEAD * NPTS * DHEAD;
  if (idx >= total) return;
  int d = idx & 31;
  int n = (idx >> 5) % NPTS;
  int h = (idx / (DHEAD * NPTS)) % NHEAD;
  int b = idx / (DHEAD * NPTS * NHEAD);
  Y[idx] = (_Float16)X[(size_t)(b * NPTS + n) * DIMC + d * NHEAD + h];
}

// [B,N,128] f32 -> [B,H,32,N] f16  (V stored d-major for contiguous PV fragments)
__global__ void k_packv(const float* __restrict__ X, _Float16* __restrict__ Y) {
  int idx = blockIdx.x * blockDim.x + threadIdx.x;
  int total = NB * NHEAD * DHEAD * NPTS;
  if (idx >= total) return;
  int n = idx % NPTS;
  int d = (idx / NPTS) & 31;
  int h = (idx / (NPTS * DHEAD)) % NHEAD;
  int b = idx / (NPTS * DHEAD * NHEAD);
  Y[idx] = (_Float16)X[(size_t)(b * NPTS + n) * DIMC + d * NHEAD + h];
}

// ---------------------------------------------------------------------------
// Flash attention: one wave = 32 queries of one (b,h); per 32-key chunk:
// 4 WMMA (S) + online softmax + LDS P-relayout + 4 WMMA (PV); K/V fragments
// shared by both query tiles.
// q,k: [B,H,N,32] f16 ; v: [B,H,32,N] f16 ; msg out: [B,N,128] f16
// ---------------------------------------------------------------------------
__global__ void k_attn(const _Float16* __restrict__ q16, const _Float16* __restrict__ k16,
                       const _Float16* __restrict__ v16p, const int* __restrict__ mask_src,
                       _Float16* __restrict__ msg, float scale) {
  const int lane = threadIdx.x;
  const int g = lane >> 4, r = lane & 15, c = lane & 15;
  const int n0 = blockIdx.x * 32, h = blockIdx.y, b = blockIdx.z;
  const _Float16* qp = q16 + (size_t)(b * NHEAD + h) * NPTS * DHEAD;
  const _Float16* kp = k16 + (size_t)(b * NHEAD + h) * NPTS * DHEAD;
  const _Float16* vp = v16p + (size_t)(b * NHEAD + h) * DHEAD * NPTS;

  v16h aq0 = frag_from(qp + (size_t)(n0 + r) * DHEAD + g * 8);
  v16h aq1 = frag_from(qp + (size_t)(n0 + 16 + r) * DHEAD + g * 8);

  float m0_[8], l0_[8], m1_[8], l1_[8];
  v8f O00 = {}, O01 = {}, O10 = {}, O11 = {};
#pragma unroll
  for (int v = 0; v < 8; ++v) { m0_[v] = -1e30f; l0_[v] = 0.f; m1_[v] = -1e30f; l1_[v] = 0.f; }

  __shared__ _Float16 sP[32][32];

#define ONLINE_UPDATE(S0, S1, M_, L_, OA, OB, ROWBASE)                          \
  {                                                                             \
    _Pragma("unroll")                                                           \
    for (int v = 0; v < 8; ++v) {                                               \
      float a_ = S0[v] * scale + msk0;                                          \
      float b_ = S1[v] * scale + msk1;                                          \
      float mv = fmaxf(a_, b_);                                                 \
      for (int w = 1; w < 16; w <<= 1) mv = fmaxf(mv, __shfl_xor(mv, w, 16));   \
      float nm = fmaxf(M_[v], mv);                                              \
      float resc = __expf(M_[v] - nm);                                          \
      float p0 = __expf(a_ - nm), p1 = __expf(b_ - nm);                         \
      float ps = p0 + p1;                                                       \
      for (int w = 1; w < 16; w <<= 1) ps += __shfl_xor(ps, w, 16);             \
      L_[v] = L_[v] * resc + ps;                                                \
      M_[v] = nm;                                                               \
      OA[v] *= resc; OB[v] *= resc;                                             \
      sP[(ROWBASE) + g * 8 + v][c] = (_Float16)p0;                              \
      sP[(ROWBASE) + g * 8 + v][16 + c] = (_Float16)p1;                         \
    }                                                                           \
  }

  for (int m0 = 0; m0 < NPTS; m0 += 32) {
    if (m0 + 32 < NPTS) {
      __builtin_prefetch((const void*)(kp + (size_t)(m0 + 32) * DHEAD), 0, 1);
      __builtin_prefetch((const void*)(vp + (size_t)c * NPTS + m0 + 32), 0, 1);
    }
    v16h bk0 = frag_from(kp + (size_t)(m0 + c) * DHEAD + g * 8);
    v16h bk1 = frag_from(kp + (size_t)(m0 + 16 + c) * DHEAD + g * 8);
    v8f z = {};
    v8f s00 = wmma_f16(aq0, bk0, z);
    v8f s01 = wmma_f16(aq0, bk1, z);
    v8f s10 = wmma_f16(aq1, bk0, z);
    v8f s11 = wmma_f16(aq1, bk1, z);
    float msk0 = mask_src[b * NPTS + m0 + c]      ? 0.f : -1e30f;
    float msk1 = mask_src[b * NPTS + m0 + 16 + c] ? 0.f : -1e30f;
    ONLINE_UPDATE(s00, s01, m0_, l0_, O00, O01, 0)
    ONLINE_UPDATE(s10, s11, m1_, l1_, O10, O11, 16)
    __syncthreads();
    v16h aP0 = frag_from(&sP[r][g * 8]);
    v16h aP1 = frag_from(&sP[16 + r][g * 8]);
    __syncthreads();
    v16h bv0 = frag_from(vp + (size_t)c * NPTS + m0 + g * 8);
    v16h bv1 = frag_from(vp + (size_t)(16 + c) * NPTS + m0 + g * 8);
    O00 = wmma_f16(aP0, bv0, O00);
    O01 = wmma_f16(aP0, bv1, O01);
    O10 = wmma_f16(aP1, bv0, O10);
    O11 = wmma_f16(aP1, bv1, O11);
  }
#undef ONLINE_UPDATE
#pragma unroll
  for (int v = 0; v < 8; ++v) {
    int na = n0 + g * 8 + v;
    int nb = n0 + 16 + g * 8 + v;
    float ia = 1.f / l0_[v], ib = 1.f / l1_[v];
    msg[(size_t)(b * NPTS + na) * DIMC + c * NHEAD + h]        = (_Float16)(O00[v] * ia);
    msg[(size_t)(b * NPTS + na) * DIMC + (16 + c) * NHEAD + h] = (_Float16)(O01[v] * ia);
    msg[(size_t)(b * NPTS + nb) * DIMC + c * NHEAD + h]        = (_Float16)(O10[v] * ib);
    msg[(size_t)(b * NPTS + nb) * DIMC + (16 + c) * NHEAD + h] = (_Float16)(O11[v] * ib);
  }
}

// cat[B,N,256] f16 = [x16 ; mrg16]
__global__ void k_concat(const _Float16* __restrict__ x, const _Float16* __restrict__ m,
                         _Float16* __restrict__ cat) {
  int idx = blockIdx.x * blockDim.x + threadIdx.x;
  int total = NB * NPTS * 256;
  if (idx >= total) return;
  int ch = idx & 255;
  int bn = idx >> 8;
  cat[idx] = (ch < DIMC) ? x[(size_t)bn * DIMC + ch] : m[(size_t)bn * DIMC + (ch - DIMC)];
}

__global__ void k_zero(float* __restrict__ p, int count) {
  int idx = blockIdx.x * blockDim.x + threadIdx.x;
  if (idx < count) p[idx] = 0.f;
}

// ---------------------------------------------------------------------------
// scores interior of coupling C: one wave = 16(n) x 64(m); md [B,N,128] f16
// software-pipelined over the 4 K-chunks
// ---------------------------------------------------------------------------
__global__ void k_scores(const _Float16* __restrict__ md0, const _Float16* __restrict__ md1,
                         float* __restrict__ C, float scale) {
  const int lane = threadIdx.x;
  const int g = lane >> 4, r = lane & 15, c = lane & 15;
  const int n0 = blockIdx.x * 16, m0 = blockIdx.y * 64, b = blockIdx.z;
  v8f acc0 = {}, acc1 = {}, acc2 = {}, acc3 = {};
  const _Float16* arow = md0 + (size_t)(b * NPTS + n0 + r) * DIMC + g * 8;
  const _Float16* b0p = md1 + (size_t)(b * NPTS + m0 + c) * DIMC + g * 8;
  const _Float16* b1p = b0p + (size_t)16 * DIMC;
  const _Float16* b2p = b0p + (size_t)32 * DIMC;
  const _Float16* b3p = b0p + (size_t)48 * DIMC;
  v16h a  = frag_from(arow);
  v16h b0 = frag_from(b0p);
  v16h b1 = frag_from(b1p);
  v16h b2 = frag_from(b2p);
  v16h b3 = frag_from(b3p);
#pragma unroll
  for (int k0 = 0; k0 < DIMC; k0 += 32) {
    v16h an, nb0, nb1, nb2, nb3;
    const bool more = (k0 + 32 < DIMC);
    if (more) {
      an  = frag_from(arow + k0 + 32);
      nb0 = frag_from(b0p + k0 + 32);
      nb1 = frag_from(b1p + k0 + 32);
      nb2 = frag_from(b2p + k0 + 32);
      nb3 = frag_from(b3p + k0 + 32);
    }
    acc0 = wmma_f16(a, b0, acc0);
    acc1 = wmma_f16(a, b1, acc1);
    acc2 = wmma_f16(a, b2, acc2);
    acc3 = wmma_f16(a, b3, acc3);
    if (more) { a = an; b0 = nb0; b1 = nb1; b2 = nb2; b3 = nb3; }
  }
  v8f accs[4] = {acc0, acc1, acc2, acc3};
#pragma unroll
  for (int j = 0; j < 4; ++j)
#pragma unroll
    for (int v = 0; v < 8; ++v) {
      int n = n0 + g * 8 + v;
      C[((size_t)b * M1 + n) * M1 + m0 + j * 16 + c] = accs[j][v] * scale;
    }
}

__global__ void k_fill_bins(float* __restrict__ C, const float* __restrict__ alpha) {
  int idx = blockIdx.x * blockDim.x + threadIdx.x;
  int total = NB * M1;
  if (idx >= total) return;
  int i = idx % M1, b = idx / M1;
  float a = alpha[0];
  C[((size_t)b * M1 + (M1 - 1)) * M1 + i] = a;
  C[((size_t)b * M1 + i) * M1 + (M1 - 1)] = a;
}

__global__ void k_sink_u(const float* __restrict__ C, const float* __restrict__ vv,
                         float* __restrict__ uu, float norm, float logN) {
  const int n = blockIdx.x, b = blockIdx.y;
  const float* row = C + ((size_t)b * M1 + n) * M1;
  const float* vb = vv + b * M1;
  __shared__ float sm[256];
  float mx = -1e30f;
  for (int m = threadIdx.x; m < M1; m += 256) mx = fmaxf(mx, row[m] + vb[m]);
  sm[threadIdx.x] = mx; __syncthreads();
  for (int off = 128; off > 0; off >>= 1) {
    if (threadIdx.x < off) sm[threadIdx.x] = fmaxf(sm[threadIdx.x], sm[threadIdx.x + off]);
    __syncthreads();
  }
  mx = sm[0]; __syncthreads();
  float s = 0.f;
  for (int m = threadIdx.x; m < M1; m += 256) s += __expf(row[m] + vb[m] - mx);
  sm[threadIdx.x] = s; __syncthreads();
  for (int off = 128; off > 0; off >>= 1) {
    if (threadIdx.x < off) sm[threadIdx.x] += sm[threadIdx.x + off];
    __syncthreads();
  }
  if (threadIdx.x == 0) {
    float lse = mx + __logf(sm[0]);
    float lmu = (n < M1 - 1) ? norm : (logN + norm);
    uu[b * M1 + n] = lmu - lse;
  }
}

__global__ void k_sink_v(const float* __restrict__ C, const float* __restrict__ uu,
                         float* __restrict__ vv, float norm, float logN) {
  const int m = blockIdx.x, b = blockIdx.y;
  const float* col = C + (size_t)b * M1 * M1 + m;
  const float* ub = uu + b * M1;
  __shared__ float sm[256];
  float mx = -1e30f;
  for (int n = threadIdx.x; n < M1; n += 256) mx = fmaxf(mx, col[(size_t)n * M1] + ub[n]);
  sm[threadIdx.x] = mx; __syncthreads();
  for (int off = 128; off > 0; off >>= 1) {
    if (threadIdx.x < off) sm[threadIdx.x] = fmaxf(sm[threadIdx.x], sm[threadIdx.x + off]);
    __syncthreads();
  }
  mx = sm[0]; __syncthreads();
  float s = 0.f;
  for (int n = threadIdx.x; n < M1; n += 256) s += __expf(col[(size_t)n * M1] + ub[n] - mx);
  sm[threadIdx.x] = s; __syncthreads();
  for (int off = 128; off > 0; off >>= 1) {
    if (threadIdx.x < off) sm[threadIdx.x] += sm[threadIdx.x + off];
    __syncthreads();
  }
  if (threadIdx.x == 0) {
    float lse = mx + __logf(sm[0]);
    float lnu = (m < M1 - 1) ? norm : (logN + norm);
    vv[b * M1 + m] = lnu - lse;
  }
}

__global__ void k_z(const float* __restrict__ C, const float* __restrict__ uu,
                    const float* __restrict__ vv, float* __restrict__ out, float norm) {
  int idx = blockIdx.x * blockDim.x + threadIdx.x;
  int total = NB * M1 * M1;
  if (idx >= total) return;
  int m = idx % M1, n = (idx / M1) % M1, b = idx / (M1 * M1);
  out[idx] = C[idx] + uu[b * M1 + n] + vv[b * M1 + m] - norm;
}

// ===========================================================================
extern "C" void kernel_launch(void* const* d_in, const int* in_sizes, int n_in,
                              void* d_out, int out_size, void* d_ws, size_t ws_size,
                              hipStream_t stream) {
  (void)in_sizes; (void)n_in; (void)out_size; (void)ws_size;
  const float* kpts0 = (const float*)d_in[0];
  const float* kpts1 = (const float*)d_in[1];
  const float* spec0 = (const float*)d_in[2];
  const float* spec1 = (const float*)d_in[3];
  const int*   mask0 = (const int*)d_in[4];
  const int*   mask1 = (const int*)d_in[5];

  int pi = 6;
  auto P = [&]() { return (const float*)d_in[pi++]; };
  const float *kW[4], *kB[4], *tW[2], *tBias[2];
  for (int i = 0; i < 4; ++i) { kW[i] = P(); kB[i] = P(); }
  for (int i = 0; i < 2; ++i) { tW[i] = P(); tBias[i] = P(); }
  struct Layer { const float *Wq,*bq,*Wk,*bk,*Wv,*bv,*Wm,*bm,*W0,*b0,*W1,*b1; } L[6];
  for (int i = 0; i < 6; ++i) {
    L[i].Wq = P(); L[i].bq = P(); L[i].Wk = P(); L[i].bk = P(); L[i].Wv = P(); L[i].bv = P();
    L[i].Wm = P(); L[i].bm = P();
    L[i].W0 = P(); L[i].b0 = P(); L[i].W1 = P(); L[i].b1 = P();
  }
  const float* Wf = P(); const float* bf = P();
  const float* alpha = P();

  // --- workspace bump allocator ---
  char* wp = (char*)d_ws;
  auto alloc = [&](size_t bytes) { void* r = (void*)wp; wp += (bytes + 255) & ~(size_t)255; return r; };
  const size_t NC128 = (size_t)NB * NPTS * DIMC;
  const size_t NC256 = (size_t)NB * NPTS * 256;
  _Float16* sp16   = (_Float16*)alloc(sizeof(_Float16) * NB * NPTS * 64);
  float*    d0_32  = (float*)alloc(sizeof(float) * NC128);
  float*    d1_32  = (float*)alloc(sizeof(float) * NC128);
  _Float16* d0_16  = (_Float16*)alloc(sizeof(_Float16) * NC128);
  _Float16* d1_16  = (_Float16*)alloc(sizeof(_Float16) * NC128);
  float*    g32    = (float*)alloc(sizeof(float) * NC256);
  _Float16* g16    = (_Float16*)alloc(sizeof(_Float16) * NC256);
  _Float16* b16    = (_Float16*)alloc(sizeof(_Float16) * NC256);
  float*    t32    = (float*)alloc(sizeof(float) * NC128);
  _Float16* junk16 = (_Float16*)alloc(sizeof(_Float16) * NC256);
  float*    h32    = (float*)alloc(sizeof(float) * NC256);
  _Float16* hn16   = (_Float16*)alloc(sizeof(_Float16) * NC256);
  _Float16* cat16  = (_Float16*)alloc(sizeof(_Float16) * NC256);
  _Float16* mrg16  = (_Float16*)alloc(sizeof(_Float16) * NC128);
  _Float16* msg16  = (_Float16*)alloc(sizeof(_Float16) * NC128);
  float*    dl0_32 = (float*)alloc(sizeof(float) * NC128);
  float*    dl1_32 = (float*)alloc(sizeof(float) * NC128);
  _Float16* md0_16 = (_Float16*)alloc(sizeof(_Float16) * NC128);
  _Float16* md1_16 = (_Float16*)alloc(sizeof(_Float16) * NC128);
  _Float16* q16 = (_Float16*)alloc(sizeof(_Float16) * NB * NHEAD * NPTS * DHEAD);
  _Float16* k16 = (_Float16*)alloc(sizeof(_Float16) * NB * NHEAD * NPTS * DHEAD);
  _Float16* v16b = (_Float16*)alloc(sizeof(_Float16) * NB * NHEAD * NPTS * DHEAD);
  float* Cc = (float*)alloc(sizeof(float) * NB * M1 * M1);
  float* uu = (float*)alloc(sizeof(float) * NB * M1);
  float* vv = (float*)alloc(sizeof(float) * NB * M1);

  auto ew = [](int count) { return dim3((count + 255) / 256); };
  auto conv = [&](const _Float16* X, const float* W, const float* bb, float* Y32,
                  _Float16* Y16, int Cin, int Cout) {
    k_conv<<<dim3(NPTS / 64, Cout / 16, NB), dim3(32), 0, stream>>>(X, W, bb, Y32, Y16, Cin, Cout);
  };
  auto inorm = [&](const float* X, _Float16* Y, int C) {
    k_inorm_relu<<<dim3(NB * C), dim3(256), 0, stream>>>(X, Y, C);
  };

  // ---------------- encoders: d = kenc(kpts) + tenc(spec) ----------------
  auto encode = [&](const float* kpts, const float* spec, float* d32, _Float16* d16) {
    k_conv2<<<ew(NB * NPTS * 32), 256, 0, stream>>>(kpts, kW[0], kB[0], g32, g16);
    inorm(g32, b16, 32);
    conv(b16, kW[1], kB[1], g32, g16, 32, 64);
    inorm(g32, b16, 64);
    conv(b16, kW[2], kB[2], g32, g16, 64, 128);
    inorm(g32, b16, 128);
    conv(b16, kW[3], kB[3], d32, junk16, 128, 128);
    k_cvt<<<ew(NB * NPTS * 64), 256, 0, stream>>>(spec, sp16, NB * NPTS * 64);
    conv(sp16, tW[0], tBias[0], g32, g16, 64, 96);
    inorm(g32, b16, 96);
    conv(b16, tW[1], tBias[1], t32, junk16, 96, 128);
    k_addpack<<<ew((int)NC128), 256, 0, stream>>>(d32, t32, d16, (int)NC128);
  };
  encode(kpts0, spec0, d0_32, d0_16);
  encode(kpts1, spec1, d1_32, d1_16);

  // ---------------- GNN layers ----------------
  const float attn_scale = 0.17677669529663687f; // 1/sqrt(32)
  const int QKV = NB * NHEAD * NPTS * DHEAD;
  auto attn_prop = [&](const _Float16* x16, const _Float16* src16, const int* msrc,
                       const Layer& lp, float* delta32) {
    conv(x16, lp.Wq, lp.bq, g32, junk16, 128, 128);
    k_packqk<<<ew(QKV), 256, 0, stream>>>(g32, q16);
    conv(src16, lp.Wk, lp.bk, g32, junk16, 128, 128);
    k_packqk<<<ew(QKV), 256, 0, stream>>>(g32, k16);
    conv(src16, lp.Wv, lp.bv, g32, junk16, 128, 128);
    k_packv<<<ew(QKV), 256, 0, stream>>>(g32, v16b);
    k_attn<<<dim3(NPTS / 32, NHEAD, NB), dim3(32), 0, stream>>>(q16, k16, v16b, msrc, msg16, attn_scale);
    conv(msg16, lp.Wm, lp.bm, g32, mrg16, 128, 128);
    k_concat<<<ew((int)NC256), 256, 0, stream>>>(x16, mrg16, cat16);
    conv(cat16, lp.W0, lp.b0, h32, junk16, 256, 256);
    inorm(h32, hn16, 256);
    conv(hn16, lp.W1, lp.b1, delta32, junk16, 256, 128);
  };
  for (int i = 0; i < 6; ++i) {
    if ((i & 1) == 0) { // self
      attn_prop(d0_16, d0_16, mask0, L[i], dl0_32);
      attn_prop(d1_16, d1_16, mask1, L[i], dl1_32);
    } else {            // cross (both deltas from pre-update d0/d1)
      attn_prop(d0_16, d1_16, mask1, L[i], dl0_32);
      attn_prop(d1_16, d0_16, mask0, L[i], dl1_32);
    }
    k_addpack<<<ew((int)NC128), 256, 0, stream>>>(d0_32, dl0_32, d0_16, (int)NC128);
    k_addpack<<<ew((int)NC128), 256, 0, stream>>>(d1_32, dl1_32, d1_16, (int)NC128);
  }

  // ---------------- final projection + scores into coupling matrix ----------------
  conv(d0_16, Wf, bf, g32, md0_16, 128, 128);
  conv(d1_16, Wf, bf, h32, md1_16, 128, 128);
  k_scores<<<dim3(NPTS / 16, NPTS / 64, NB), dim3(32), 0, stream>>>(
      md0_16, md1_16, Cc, 0.08838834764831845f /* 1/sqrt(128) */);
  k_fill_bins<<<ew(NB * M1), 256, 0, stream>>>(Cc, alpha);

  // ---------------- Sinkhorn (20 iters, log domain) ----------------
  k_zero<<<ew(NB * M1), 256, 0, stream>>>(uu, NB * M1);
  k_zero<<<ew(NB * M1), 256, 0, stream>>>(vv, NB * M1);
  const float norm = -logf((float)(NPTS + NPTS));
  const float logN = logf((float)NPTS);
  for (int it = 0; it < 20; ++it) {
    k_sink_u<<<dim3(M1, NB), 256, 0, stream>>>(Cc, vv, uu, norm, logN);
    k_sink_v<<<dim3(M1, NB), 256, 0, stream>>>(Cc, uu, vv, norm, logN);
  }
  k_z<<<ew(NB * M1 * M1), 256, 0, stream>>>(Cc, uu, vv, (float*)d_out, norm);
}
